// Model_41274635714613
// MI455X (gfx1250) — matrix operand order repaired and checked by reading, not compiled, not run on hardware
//
#include <hip/hip_runtime.h>
#include <hip/hip_bf16.h>
#include <math.h>

// ---------------------------------------------------------------------------
// Types / helpers
// ---------------------------------------------------------------------------
typedef __attribute__((ext_vector_type(16))) __bf16 v16bf;
typedef __attribute__((ext_vector_type(8)))  float  v8f;

#define DEVINL __device__ __forceinline__
#define GS_LOOP(i, n) for (long long i = (long long)blockIdx.x * blockDim.x + threadIdx.x; \
                           i < (n); i += (long long)gridDim.x * blockDim.x)

DEVINL unsigned short f2bf(float f) {
  unsigned int u = __float_as_uint(f);
  u += 0x7fffu + ((u >> 16) & 1u);          // round-to-nearest-even
  return (unsigned short)(u >> 16);
}
DEVINL unsigned int fenc(float f) {          // order-preserving float -> uint
  unsigned int u = __float_as_uint(f);
  return (u & 0x80000000u) ? ~u : (u | 0x80000000u);
}
DEVINL float fdec(unsigned int u) {
  return __uint_as_float((u & 0x80000000u) ? (u & 0x7fffffffu) : ~u);
}
DEVINL float sigf(float x) { return 1.0f / (1.0f + expf(-x)); }
DEVINL void edge_sd(const int* ei, long long E, long long e, int& s, int& d) {
  if (e < E) { s = ei[e]; d = ei[E + e]; }
  else       { int n = (int)(e - E); s = n; d = n; }   // appended self-loops
}

// ---------------------------------------------------------------------------
// Fills
// ---------------------------------------------------------------------------
__global__ void fill_f32(float* p, float v, long long n) { GS_LOOP(i, n) p[i] = v; }
__global__ void fill_u32(unsigned int* p, unsigned int v, long long n) { GS_LOOP(i, n) p[i] = v; }

// ---------------------------------------------------------------------------
// Packing: f32 -> padded bf16 (activations, weights transposed / direct)
// ---------------------------------------------------------------------------
__global__ void pack_a_k(const float* __restrict__ S, unsigned short* __restrict__ D,
                         int K, int lds, int Kp, long long tot) {
  GS_LOOP(i, tot) {
    long long m = i / Kp; int k = (int)(i % Kp);
    D[i] = (k < K) ? f2bf(S[m * lds + k]) : (unsigned short)0;
  }
}
// W stored [K, N] row-major (torch Linear style in the reference: (in, out))
__global__ void pack_w_t_k(const float* __restrict__ W, unsigned short* __restrict__ D,
                           int K, int N, int Kp, int Np, long long tot) {
  GS_LOOP(i, tot) {
    int n = (int)(i / Kp); int k = (int)(i % Kp);
    D[i] = (k < K && n < N) ? f2bf(W[(size_t)k * N + n]) : (unsigned short)0;
  }
}
// W stored [N, K] row-major (LSTM wih: (4h, in))
__global__ void pack_w_d_k(const float* __restrict__ W, unsigned short* __restrict__ D,
                           int K, int N, int Kp, int Np, long long tot) {
  GS_LOOP(i, tot) {
    int n = (int)(i / Kp); int k = (int)(i % Kp);
    D[i] = (k < K && n < N) ? f2bf(W[(size_t)n * K + k]) : (unsigned short)0;
  }
}

// ---------------------------------------------------------------------------
// WMMA GEMM: C[M, ntn*16] = A[M, Kpad] * Bt[Npad, Kpad]^T  (+bias, +relu)
// Register-blocked: one wave computes a 32x32 super-tile (2x2 of 16x16 WMMA
// tiles), 4x v_wmma_f32_16x16x32_bf16 per K-step, f32 accumulate.
// Fragment layouts per CDNA5 ISA 7.12.2 (16-bit A 16x32, B 32x16, f32 C 16x16).
// M must be a multiple of 32; odd N-tile counts handled by clamping the
// second N-tile (duplicate compute, guarded store).
// ---------------------------------------------------------------------------
__global__ void gemm_bf16_wmma(const unsigned short* __restrict__ A,
                               const unsigned short* __restrict__ Bt,
                               const float* __restrict__ bias,
                               float* __restrict__ C,
                               int M, int N, int Kpad, int relu) {
  const int ntn  = (N + 15) >> 4;      // 16-wide N tiles (C row stride = ntn*16)
  const int ntn2 = (ntn + 1) >> 1;     // 32-wide super-tiles in N
  const int ntm2 = M >> 5;             // 32-wide super-tiles in M
  const long long tiles = (long long)ntm2 * ntn2;
  const long long wid = (long long)blockIdx.x * (blockDim.x >> 5) + (threadIdx.x >> 5);
  if (wid >= tiles) return;
  const int tm2 = (int)(wid / ntn2);
  const int tn2 = (int)(wid % ntn2);
  const int lane = threadIdx.x & 31;
  const int half = lane >> 4;          // which 16-lane half
  const int l15  = lane & 15;
  const int ldc  = ntn << 4;

  const int tnA = tn2 * 2;
  const int tnB = (tnA + 1 < ntn) ? (tnA + 1) : tnA;   // clamp at odd edge

  const unsigned short* arow0 = A  + (size_t)(tm2 * 32 + l15) * Kpad;
  const unsigned short* arow1 = arow0 + (size_t)16 * Kpad;
  const unsigned short* brow0 = Bt + (size_t)(tnA * 16 + l15) * Kpad;
  const unsigned short* brow1 = Bt + (size_t)(tnB * 16 + l15) * Kpad;

  union FragU { unsigned int u[8]; v16bf v; };
  v8f acc00 = {}, acc01 = {}, acc10 = {}, acc11 = {};
  for (int kk = 0; kk < Kpad; kk += 32) {
    FragU a0, a1, b0, b1;
    const unsigned int* ap0 = (const unsigned int*)(arow0 + kk);
    const unsigned int* ap1 = (const unsigned int*)(arow1 + kk);
    const unsigned int* bp0 = (const unsigned int*)(brow0 + kk);
    const unsigned int* bp1 = (const unsigned int*)(brow1 + kk);
    __builtin_prefetch(arow0 + kk + 256, 0, 1);   // global_prefetch next A chunks
    __builtin_prefetch(arow1 + kk + 256, 0, 1);
#pragma unroll
    for (int j = 0; j < 4; ++j) {
      // A frag: lanes0-15 K=0..7/16..23 ; lanes16-31 K=8..15/24..31 (bf16 pairs)
      a0.u[j]     = ap0[half * 4 + j];
      a0.u[4 + j] = ap0[8 + half * 4 + j];
      a1.u[j]     = ap1[half * 4 + j];
      a1.u[4 + j] = ap1[8 + half * 4 + j];
    }
#pragma unroll
    for (int j = 0; j < 8; ++j) {
      // B frag: lanes0-15 K=0..7/16..23 ; lanes16-31 K=8..15/24..31 (col = lane&15, Bt [N,K])
      b0.u[j] = bp0[(j >> 2) * 8 + half * 4 + (j & 3)];
      b1.u[j] = bp1[(j >> 2) * 8 + half * 4 + (j & 3)];
    }
    acc00 = __builtin_amdgcn_wmma_f32_16x16x32_bf16(false, a0.v, false, b0.v, (short)0, acc00, false, false);
    acc01 = __builtin_amdgcn_wmma_f32_16x16x32_bf16(false, a0.v, false, b1.v, (short)0, acc01, false, false);
    acc10 = __builtin_amdgcn_wmma_f32_16x16x32_bf16(false, a1.v, false, b0.v, (short)0, acc10, false, false);
    acc11 = __builtin_amdgcn_wmma_f32_16x16x32_bf16(false, a1.v, false, b1.v, (short)0, acc11, false, false);
  }
  const int n0 = (tnA << 4) + l15;
  const int n1 = (tnB << 4) + l15;
  const float bv0 = (bias != nullptr && n0 < N) ? bias[n0] : 0.0f;
  const float bv1 = (bias != nullptr && n1 < N) ? bias[n1] : 0.0f;
  const int storeB = (tnB != tnA);
#pragma unroll
  for (int j = 0; j < 8; ++j) {
    int m0 = (tm2 << 5) + j + half * 8;   // C layout: VGPR j -> M=j (+8 hi half)
    int m1 = m0 + 16;
    float v00 = acc00[j] + bv0, v01 = acc01[j] + bv1;
    float v10 = acc10[j] + bv0, v11 = acc11[j] + bv1;
    if (relu) {
      v00 = fmaxf(v00, 0.0f); v01 = fmaxf(v01, 0.0f);
      v10 = fmaxf(v10, 0.0f); v11 = fmaxf(v11, 0.0f);
    }
    C[(size_t)m0 * ldc + n0] = v00;
    C[(size_t)m1 * ldc + n0] = v10;
    if (storeB) {
      C[(size_t)m0 * ldc + n1] = v01;
      C[(size_t)m1 * ldc + n1] = v11;
    }
  }
}

// ---------------------------------------------------------------------------
// GAT attention
// ---------------------------------------------------------------------------
__global__ void gat_scores_k(const float* __restrict__ h, const float* __restrict__ asrc,
                             const float* __restrict__ adst, float* __restrict__ es,
                             float* __restrict__ ed, long long tot) {   // tot = N*10
  GS_LOOP(i, tot) {
    long long n = i / 10; int hd = (int)(i % 10);
    const float* hr = h + n * 784 + hd * 78;
    const float* as = asrc + hd * 78;
    const float* ad = adst + hd * 78;
    float s = 0.f, d = 0.f;
    for (int f = 0; f < 78; ++f) { s += hr[f] * as[f]; d += hr[f] * ad[f]; }
    es[i] = s; ed[i] = d;
  }
}
DEVINL float gat_e(const float* es, const float* ed, int s, int d, int hd) {
  float v = es[s * 10 + hd] + ed[d * 10 + hd];
  return v > 0.0f ? v : 0.2f * v;                       // leaky_relu(0.2)
}
__global__ void gat_max_k(const int* ei, const float* es, const float* ed,
                          unsigned int* mb, long long E, long long tot) {
  GS_LOOP(i, tot) {
    long long e = i / 10; int hd = (int)(i % 10); int s, d;
    edge_sd(ei, E, e, s, d);
    atomicMax(&mb[(size_t)d * 10 + hd], fenc(gat_e(es, ed, s, d, hd)));
  }
}
__global__ void gat_den_k(const int* ei, const float* es, const float* ed,
                          const unsigned int* mb, float* den, long long E, long long tot) {
  GS_LOOP(i, tot) {
    long long e = i / 10; int hd = (int)(i % 10); int s, d;
    edge_sd(ei, E, e, s, d);
    float ex = expf(gat_e(es, ed, s, d, hd) - fdec(mb[(size_t)d * 10 + hd]));
    atomicAdd(&den[(size_t)d * 10 + hd], ex);
  }
}
__global__ void gat_scatter_k(const int* ei, const float* es, const float* ed,
                              const unsigned int* mb, const float* __restrict__ h,
                              float* acc, long long E, long long tot) {
  GS_LOOP(i, tot) {
    long long e = i / 10; int hd = (int)(i % 10); int s, d;
    edge_sd(ei, E, e, s, d);
    float ex = expf(gat_e(es, ed, s, d, hd) - fdec(mb[(size_t)d * 10 + hd]));
    const float* hr = h + (size_t)s * 784 + hd * 78;
    float* ar = acc + (size_t)d * 784 + hd * 78;
    for (int f = 0; f < 78; ++f) atomicAdd(&ar[f], hr[f] * ex);
  }
}
__global__ void gat_final_k(const float* acc, const float* den, const float* gb,
                            float* xo, long long tot) {   // tot = N*780
  GS_LOOP(i, tot) {
    long long m = i / 780; int c = (int)(i % 780); int hd = c / 78;
    float v = acc[m * 784 + c] / den[m * 10 + hd] + gb[c];
    xo[m * 784 + c] = fmaxf(v, 0.0f);
  }
}

// ---------------------------------------------------------------------------
// GCN normalization + scatter
// ---------------------------------------------------------------------------
__global__ void deg_k(const int* ei, float* deg, long long E, long long tot) {
  GS_LOOP(i, tot) { int s, d; edge_sd(ei, E, i, s, d); atomicAdd(&deg[d], 1.0f); }
}
__global__ void dinv_k(float* deg, long long n) {
  GS_LOOP(i, n) deg[i] = rsqrtf(fmaxf(deg[i], 1.0f));
}
__global__ void gcn_scatter_k(const float* __restrict__ h, const int* ei,
                              const float* __restrict__ dinv, float* acc,
                              long long E, long long tot) {   // tot = ETOT*780
  GS_LOOP(i, tot) {
    long long e = i / 780; int f = (int)(i % 780); int s, d;
    edge_sd(ei, E, e, s, d);
    atomicAdd(&acc[(size_t)d * 784 + f], h[(size_t)s * 784 + f] * dinv[s] * dinv[d]);
  }
}
// x = relu(acc + bias); r = x + prev
__global__ void bias_relu_res_k(const float* acc, const float* bias, const float* prev,
                                float* xo, float* ro, long long tot) {   // tot = N*780
  GS_LOOP(i, tot) {
    long long m = i / 780; int f = (int)(i % 780);
    size_t idx = m * 784 + f;
    float x = fmaxf(acc[idx] + bias[f], 0.0f);
    xo[idx] = x;
    ro[idx] = x + prev[idx];
  }
}

// ---------------------------------------------------------------------------
// Segment pooling (max | mean) -> [B, 1568] (780 max, 780 mean, 8 zero pad)
// ---------------------------------------------------------------------------
__global__ void cnt_k(const int* batch, float* cnt, long long n) {
  GS_LOOP(i, n) atomicAdd(&cnt[batch[i]], 1.0f);
}
__global__ void pool_scatter_k(const float* __restrict__ r, const int* batch,
                               float* pool, long long tot) {   // tot = N*780
  GS_LOOP(i, tot) {
    long long n = i / 780; int f = (int)(i % 780);
    int g = batch[n];
    float v = r[n * 784 + f];
    atomicMax((unsigned int*)(pool + (size_t)g * 1568 + f), fenc(v));
    atomicAdd(pool + (size_t)g * 1568 + 780 + f, v);
  }
}
__global__ void pool_final_k(float* pool, const float* cnt, long long tot) { // B*780
  GS_LOOP(i, tot) {
    long long g = i / 780; int f = (int)(i % 780);
    size_t base = g * 1568;
    pool[base + f] = fdec(__float_as_uint(pool[base + f]));
    pool[base + 780 + f] /= cnt[g];
  }
}
__global__ void combine_pg_k(const float* p1, const float* p2, const float* p3,
                             const float* w3, float* pg, long long tot) {   // B*128
  GS_LOOP(i, tot) {
    float a = w3[0], b = w3[1], c = w3[2];
    float m = fmaxf(a, fmaxf(b, c));
    float e0 = expf(a - m), e1 = expf(b - m), e2 = expf(c - m);
    float s = e0 + e1 + e2;
    pg[i] = (p1[i] * e0 + p2[i] * e1 + p3[i] * e2) / s;
  }
}

// ---------------------------------------------------------------------------
// SMILES branch
// ---------------------------------------------------------------------------
__global__ void embed_pack_k(const int* comp, const float* emb, unsigned short* D,
                             long long tot) {   // tot = 61440*224
  GS_LOOP(i, tot) {
    long long bt = i / 224; int d = (int)(i % 224);
    D[i] = (d < 200) ? f2bf(emb[(size_t)comp[bt] * 200 + d]) : (unsigned short)0;
  }
}
// Persistent single-workgroup LSTM recurrence; xproj = x@Wih^T + b precomputed.
// h kept in LDS; c distributed across threads. out[(t*Nb+nb)*out_ld + off + hx] = h.
__global__ void lstm_seq_k(const float* __restrict__ xproj, const float* __restrict__ whh,
                           int T, int Nb, int H, int reverse,
                           float* __restrict__ out, int out_ld, int out_off) {
  __shared__ float sh[15360];                      // Nb*H <= 120*128, 60 KB
  const int G = 4 * H;
  const int cells = Nb * H;
  const int tid = threadIdx.x;
  float creg[16], hnew[16];
#pragma unroll
  for (int k = 0; k < 16; ++k) creg[k] = 0.0f;
  for (int c = tid; c < cells; c += blockDim.x) sh[c] = 0.0f;
  __syncthreads();
  for (int t = 0; t < T; ++t) {
    const int te = reverse ? (T - 1 - t) : t;
    const float* xp = xproj + (size_t)te * Nb * G;
    int idx = 0;
    for (int c = tid; c < cells; c += blockDim.x, ++idx) {
      const int nb = c / H, hx = c % H;
      const float* hr = sh + nb * H;
      const float* xr = xp + (size_t)nb * G;
      float gi = xr[hx], gf = xr[H + hx], gg = xr[2 * H + hx], go = xr[3 * H + hx];
      const float* wi = whh + (size_t)hx * H;
      const float* wf = whh + (size_t)(H + hx) * H;
      const float* wg = whh + (size_t)(2 * H + hx) * H;
      const float* wo = whh + (size_t)(3 * H + hx) * H;
      for (int k = 0; k < H; ++k) {
        float hv = hr[k];
        gi += wi[k] * hv; gf += wf[k] * hv; gg += wg[k] * hv; go += wo[k] * hv;
      }
      float cn = sigf(gf) * creg[idx] + sigf(gi) * tanhf(gg);
      creg[idx] = cn;
      hnew[idx] = sigf(go) * tanhf(cn);
    }
    __syncthreads();
    idx = 0;
    for (int c = tid; c < cells; c += blockDim.x, ++idx) {
      sh[c] = hnew[idx];
      const int nb = c / H, hx = c % H;
      out[((size_t)te * Nb + nb) * out_ld + out_off + hx] = hnew[idx];
    }
    __syncthreads();
  }
}
__global__ void mean_l_k(const float* lo, float* xdpre, long long tot) {   // 512*128
  GS_LOOP(i, tot) {
    long long b = i / 128; int d = (int)(i % 128);
    float s = 0.f;
    for (int l = 0; l < 120; ++l) s += lo[((size_t)b * 120 + l) * 128 + d];
    xdpre[i] = s / 120.0f;
  }
}
__global__ void pool78_k(const float* xdpre, float* xd, long long tot) {   // 512*78
  GS_LOOP(i, tot) {
    long long b = i / 78; int j = (int)(i % 78);
    int s = (j * 128) / 78;
    int e = ((j + 1) * 128 + 77) / 78;
    float acc = 0.f;
    for (int k = s; k < e; ++k) acc += xdpre[b * 128 + k];
    xd[i] = acc / (float)(e - s);
  }
}
__global__ void assemble_xc_k(const float* pg, const float* xd, const float* tgt,
                              unsigned short* D, long long tot) {   // 512*1792
  GS_LOOP(i, tot) {
    long long b = i / 1792; int k = (int)(i % 1792);
    float v;
    if (k < 128)        v = pg[b * 128 + k];
    else if (k < 206)   v = xd[b * 78 + (k - 128)];
    else if (k < 1768)  v = tgt[b * 1562 + (k - 206)];
    else                v = 0.0f;
    D[i] = f2bf(v);
  }
}
__global__ void copy_out_k(const float* s, float* o, long long n) {
  GS_LOOP(i, n) o[i] = s[i * 16];
}

// ---------------------------------------------------------------------------
// Host orchestration
// ---------------------------------------------------------------------------
static dim3 gsz(long long n, int bs = 256) {
  long long b = (n + bs - 1) / bs;
  if (b < 1) b = 1;
  if (b > 524288LL) b = 524288LL;
  return dim3((unsigned)b);
}

extern "C" void kernel_launch(void* const* d_in, const int* in_sizes, int n_in,
                              void* d_out, int out_size, void* d_ws, size_t ws_size,
                              hipStream_t stream) {
  (void)in_sizes; (void)n_in; (void)out_size; (void)ws_size;

  const int   Nn = 51200; const long long Ev = 262144, ETOT = 313344;
  const int   Bg = 512, Tt = 512, Ll = 120, BT = 61440;

  const float* X     = (const float*)d_in[0];
  const int*   EI    = (const int*)d_in[1];
  const int*   BATCH = (const int*)d_in[2];
  const int*   COMP  = (const int*)d_in[3];
  const float* TGT   = (const float*)d_in[4];
  const float* GATW = (const float*)d_in[5],  *ASRC = (const float*)d_in[6];
  const float* ADST = (const float*)d_in[7],  *GATB = (const float*)d_in[8];
  const float* G1W = (const float*)d_in[9],   *G1B = (const float*)d_in[10];
  const float* G2W = (const float*)d_in[11],  *G2B = (const float*)d_in[12];
  const float* G3W = (const float*)d_in[13],  *G3B = (const float*)d_in[14];
  const float* FUW = (const float*)d_in[15],  *FUB = (const float*)d_in[16];
  const float* F11W = (const float*)d_in[17], *F11B = (const float*)d_in[18];
  const float* F12W = (const float*)d_in[19], *F12B = (const float*)d_in[20];
  const float* F13W = (const float*)d_in[21], *F13B = (const float*)d_in[22];
  const float* F21W = (const float*)d_in[23], *F21B = (const float*)d_in[24];
  const float* F22W = (const float*)d_in[25], *F22B = (const float*)d_in[26];
  const float* F23W = (const float*)d_in[27], *F23B = (const float*)d_in[28];
  const float* W3   = (const float*)d_in[29];
  const float* EMB  = (const float*)d_in[30];
  const float* BFWIH = (const float*)d_in[31], *BFWHH = (const float*)d_in[32], *BFB = (const float*)d_in[33];
  const float* BBWIH = (const float*)d_in[34], *BBWHH = (const float*)d_in[35], *BBB = (const float*)d_in[36];
  const float* LWIH  = (const float*)d_in[37], *LWHH  = (const float*)d_in[38], *LB  = (const float*)d_in[39];
  const float* FC1W = (const float*)d_in[40], *FC1B = (const float*)d_in[41];
  const float* FC2W = (const float*)d_in[42], *FC2B = (const float*)d_in[43];
  const float* FC3W = (const float*)d_in[44], *FC3B = (const float*)d_in[45];
  const float* FC4W = (const float*)d_in[46], *FC4B = (const float*)d_in[47];
  const float* OW   = (const float*)d_in[48], *OB   = (const float*)d_in[49];
  float* OUT = (float*)d_out;

  // ---- workspace carve-out ------------------------------------------------
  char* ws = (char*)d_ws; size_t cur = 0;
  auto alloc = [&](size_t bytes) -> char* {
    size_t o = cur; cur += (bytes + 255) & ~(size_t)255; return ws + o;
  };
  float* hbuf = (float*)alloc((size_t)Nn * 784 * 4);
  float* accb = (float*)alloc((size_t)Nn * 784 * 4);
  float* nb0  = (float*)alloc((size_t)Nn * 784 * 4);
  float* nb1  = (float*)alloc((size_t)Nn * 784 * 4);
  float* nb2  = (float*)alloc((size_t)Nn * 784 * 4);
  unsigned short* abf = (unsigned short*)alloc((size_t)Nn * 800 * 2);
  unsigned short* xbf = (unsigned short*)alloc((size_t)Nn * 96 * 2);
  float* es  = (float*)alloc((size_t)Nn * 10 * 4);
  float* ed  = (float*)alloc((size_t)Nn * 10 * 4);
  float* den = (float*)alloc((size_t)Nn * 10 * 4);
  unsigned int* mb = (unsigned int*)alloc((size_t)Nn * 10 * 4);
  float* deg = (float*)alloc((size_t)Nn * 4);
  float* cnt = (float*)alloc((size_t)Bg * 4);
  float* poolf = (float*)alloc((size_t)Bg * 1568 * 4);
  unsigned short* poolbf = (unsigned short*)alloc((size_t)Bg * 1568 * 2);
  float* g1out = (float*)alloc((size_t)Bg * 1504 * 4);
  unsigned short* g1bf = (unsigned short*)alloc((size_t)Bg * 1504 * 2);
  float* p1 = (float*)alloc((size_t)Bg * 128 * 4);
  float* p2 = (float*)alloc((size_t)Bg * 128 * 4);
  float* p3 = (float*)alloc((size_t)Bg * 128 * 4);
  float* pg = (float*)alloc((size_t)Bg * 128 * 4);
  unsigned short* embp = (unsigned short*)alloc((size_t)BT * 224 * 2);
  float* xpF = (float*)alloc((size_t)BT * 256 * 4);
  float* xpB = (float*)alloc((size_t)BT * 256 * 4);
  float* bif = (float*)alloc((size_t)BT * 128 * 4);
  unsigned short* bibf = (unsigned short*)alloc((size_t)BT * 128 * 2);
  float* xp2 = (float*)alloc((size_t)BT * 512 * 4);
  float* lo  = (float*)alloc((size_t)BT * 128 * 4);
  float* xdpre = (float*)alloc((size_t)Bg * 128 * 4);
  float* xd    = (float*)alloc((size_t)Bg * 78 * 4);
  unsigned short* xcbf = (unsigned short*)alloc((size_t)Bg * 1792 * 2);   // reused as MLP act bf
  float* mlpA = (float*)alloc((size_t)Bg * 1024 * 4);
  float* mlpB = (float*)alloc((size_t)Bg * 1024 * 4);
  float* out16 = (float*)alloc((size_t)Bg * 16 * 4);
  unsigned short* wGat = (unsigned short*)alloc((size_t)784 * 96 * 2);
  unsigned short* wG1  = (unsigned short*)alloc((size_t)784 * 800 * 2);
  unsigned short* wG2  = (unsigned short*)alloc((size_t)784 * 800 * 2);
  unsigned short* wG3  = (unsigned short*)alloc((size_t)784 * 800 * 2);
  unsigned short* wFu  = (unsigned short*)alloc((size_t)784 * 800 * 2);
  unsigned short* wF11 = (unsigned short*)alloc((size_t)1504 * 1568 * 2);
  unsigned short* wF12 = (unsigned short*)alloc((size_t)1504 * 1568 * 2);
  unsigned short* wF13 = (unsigned short*)alloc((size_t)1504 * 1568 * 2);
  unsigned short* wF21 = (unsigned short*)alloc((size_t)128 * 1504 * 2);
  unsigned short* wF22 = (unsigned short*)alloc((size_t)128 * 1504 * 2);
  unsigned short* wF23 = (unsigned short*)alloc((size_t)128 * 1504 * 2);
  unsigned short* wIF  = (unsigned short*)alloc((size_t)256 * 224 * 2);
  unsigned short* wIB  = (unsigned short*)alloc((size_t)256 * 224 * 2);
  unsigned short* wI2  = (unsigned short*)alloc((size_t)512 * 128 * 2);
  unsigned short* wC1  = (unsigned short*)alloc((size_t)1024 * 1792 * 2);
  unsigned short* wC2  = (unsigned short*)alloc((size_t)1024 * 1024 * 2);
  unsigned short* wC3  = (unsigned short*)alloc((size_t)512 * 1024 * 2);
  unsigned short* wC4  = (unsigned short*)alloc((size_t)256 * 512 * 2);
  unsigned short* wO   = (unsigned short*)alloc((size_t)16 * 256 * 2);

  // ---- launch helpers -----------------------------------------------------
  auto packWT = [&](const float* W, unsigned short* D, int K, int N, int Kp, int Np) {
    long long tot = (long long)Np * Kp;
    pack_w_t_k<<<gsz(tot), 256, 0, stream>>>(W, D, K, N, Kp, Np, tot);
  };
  auto packWD = [&](const float* W, unsigned short* D, int K, int N, int Kp, int Np) {
    long long tot = (long long)Np * Kp;
    pack_w_d_k<<<gsz(tot), 256, 0, stream>>>(W, D, K, N, Kp, Np, tot);
  };
  auto packA = [&](const float* S, unsigned short* D, int M, int K, int lds, int Kp) {
    long long tot = (long long)M * Kp;
    pack_a_k<<<gsz(tot), 256, 0, stream>>>(S, D, K, lds, Kp, tot);
  };
  auto gemm = [&](const unsigned short* A, const unsigned short* Bt, const float* bias,
                  int relu, float* C, int M, int N, int Kp) {
    int ntn = (N + 15) >> 4;
    int ntn2 = (ntn + 1) >> 1;
    long long tiles = (long long)(M >> 5) * ntn2;   // 32x32 super-tiles, M % 32 == 0
    int blocks = (int)((tiles + 3) / 4);
    gemm_bf16_wmma<<<blocks, 128, 0, stream>>>(A, Bt, bias, C, M, N, Kp, relu);
  };
  auto zf = [&](float* p, long long n) { fill_f32<<<gsz(n), 256, 0, stream>>>(p, 0.f, n); };
  auto zu = [&](unsigned int* p, long long n) { fill_u32<<<gsz(n), 256, 0, stream>>>(p, 0u, n); };
  auto doPool = [&](const float* r, float* pout, unsigned short* wA, const float* bA,
                    unsigned short* wB, const float* bB) {
    zf(poolf, (long long)Bg * 1568);
    pool_scatter_k<<<gsz((long long)Nn * 780), 256, 0, stream>>>(r, BATCH, poolf, (long long)Nn * 780);
    pool_final_k<<<gsz((long long)Bg * 780), 256, 0, stream>>>(poolf, cnt, (long long)Bg * 780);
    packA(poolf, poolbf, Bg, 1568, 1568, 1568);
    gemm(poolbf, wA, bA, 1, g1out, Bg, 1500, 1568);
    packA(g1out, g1bf, Bg, 1504, 1504, 1504);
    gemm(g1bf, wB, bB, 0, pout, Bg, 128, 1504);
  };

  // ---- weight packs (independent) ----------------------------------------
  packWT(GATW, wGat, 78, 780, 96, 784);
  packWT(G1W, wG1, 780, 780, 800, 784);
  packWT(G2W, wG2, 780, 780, 800, 784);
  packWT(G3W, wG3, 780, 780, 800, 784);
  packWT(FUW, wFu, 780, 780, 800, 784);
  packWT(F11W, wF11, 1560, 1500, 1568, 1504);
  packWT(F12W, wF12, 1560, 1500, 1568, 1504);
  packWT(F13W, wF13, 1560, 1500, 1568, 1504);
  packWT(F21W, wF21, 1500, 128, 1504, 128);
  packWT(F22W, wF22, 1500, 128, 1504, 128);
  packWT(F23W, wF23, 1500, 128, 1504, 128);
  packWD(BFWIH, wIF, 200, 256, 224, 256);
  packWD(BBWIH, wIB, 200, 256, 224, 256);
  packWD(LWIH, wI2, 128, 512, 128, 512);
  packWT(FC1W, wC1, 1768, 1024, 1792, 1024);
  packWT(FC2W, wC2, 1024, 1024, 1024, 1024);
  packWT(FC3W, wC3, 1024, 512, 1024, 512);
  packWT(FC4W, wC4, 512, 256, 512, 256);
  packWT(OW, wO, 256, 1, 256, 16);

  // ---- GAT ----------------------------------------------------------------
  packA(X, xbf, Nn, 78, 78, 96);
  gemm(xbf, wGat, nullptr, 0, hbuf, Nn, 780, 96);                 // h = x @ gat_w
  gat_scores_k<<<gsz((long long)Nn * 10), 256, 0, stream>>>(hbuf, ASRC, ADST, es, ed, (long long)Nn * 10);
  zu(mb, (long long)Nn * 10);
  zf(den, (long long)Nn * 10);
  zf(accb, (long long)Nn * 784);
  gat_max_k<<<gsz(ETOT * 10), 256, 0, stream>>>(EI, es, ed, mb, Ev, ETOT * 10);
  gat_den_k<<<gsz(ETOT * 10), 256, 0, stream>>>(EI, es, ed, mb, den, Ev, ETOT * 10);
  gat_scatter_k<<<gsz(ETOT * 10), 256, 0, stream>>>(EI, es, ed, mb, hbuf, accb, Ev, ETOT * 10);
  gat_final_k<<<gsz((long long)Nn * 780), 256, 0, stream>>>(accb, den, GATB, nb0, (long long)Nn * 780);

  // degrees / pool counts (shared across layers)
  zf(deg, Nn);
  deg_k<<<gsz(ETOT), 256, 0, stream>>>(EI, deg, Ev, ETOT);
  dinv_k<<<gsz(Nn), 256, 0, stream>>>(deg, Nn);
  zf(cnt, Bg);
  cnt_k<<<gsz(Nn), 256, 0, stream>>>(BATCH, cnt, Nn);

  // ---- GCN1 + pool1 -------------------------------------------------------
  packA(nb0, abf, Nn, 780, 784, 800);
  gemm(abf, wG1, nullptr, 0, hbuf, Nn, 780, 800);
  zf(accb, (long long)Nn * 784);
  gcn_scatter_k<<<gsz(ETOT * 780), 256, 0, stream>>>(hbuf, EI, deg, accb, Ev, ETOT * 780);
  bias_relu_res_k<<<gsz((long long)Nn * 780), 256, 0, stream>>>(accb, G1B, nb0, nb1, nb2, (long long)Nn * 780);
  doPool(nb2, p1, wF11, F11B, wF21, F21B);

  // ---- GCN2 + pool2 -------------------------------------------------------
  packA(nb2, abf, Nn, 780, 784, 800);
  gemm(abf, wG2, nullptr, 0, hbuf, Nn, 780, 800);
  zf(accb, (long long)Nn * 784);
  gcn_scatter_k<<<gsz(ETOT * 780), 256, 0, stream>>>(hbuf, EI, deg, accb, Ev, ETOT * 780);
  bias_relu_res_k<<<gsz((long long)Nn * 780), 256, 0, stream>>>(accb, G2B, nb1, nb0, nb2, (long long)Nn * 780);
  doPool(nb2, p2, wF12, F12B, wF22, F22B);

  // ---- GCN3 + fusion + pool3 ---------------------------------------------
  packA(nb2, abf, Nn, 780, 784, 800);
  gemm(abf, wG3, nullptr, 0, hbuf, Nn, 780, 800);
  zf(accb, (long long)Nn * 784);
  gcn_scatter_k<<<gsz(ETOT * 780), 256, 0, stream>>>(hbuf, EI, deg, accb, Ev, ETOT * 780);
  bias_relu_res_k<<<gsz((long long)Nn * 780), 256, 0, stream>>>(accb, G3B, nb0, nb1, nb2, (long long)Nn * 780);
  packA(nb2, abf, Nn, 780, 784, 800);
  gemm(abf, wFu, FUB, 1, hbuf, Nn, 780, 800);                     // fusion = relu(r3@W+b)
  doPool(hbuf, p3, wF13, F13B, wF23, F23B);

  combine_pg_k<<<gsz((long long)Bg * 128), 256, 0, stream>>>(p1, p2, p3, W3, pg, (long long)Bg * 128);

  // ---- SMILES branch ------------------------------------------------------
  embed_pack_k<<<gsz((long long)BT * 224), 256, 0, stream>>>(COMP, EMB, embp, (long long)BT * 224);
  gemm(embp, wIF, BFB, 0, xpF, BT, 256, 224);
  gemm(embp, wIB, BBB, 0, xpB, BT, 256, 224);
  lstm_seq_k<<<1, 1024, 0, stream>>>(xpF, BFWHH, Tt, Ll, 64, 0, bif, 128, 0);
  lstm_seq_k<<<1, 1024, 0, stream>>>(xpB, BBWHH, Tt, Ll, 64, 1, bif, 128, 64);
  packA(bif, bibf, BT, 128, 128, 128);
  gemm(bibf, wI2, LB, 0, xp2, BT, 512, 128);
  lstm_seq_k<<<1, 1024, 0, stream>>>(xp2, LWHH, Tt, Ll, 128, 0, lo, 128, 0);
  mean_l_k<<<gsz((long long)Bg * 128), 256, 0, stream>>>(lo, xdpre, (long long)Bg * 128);
  pool78_k<<<gsz((long long)Bg * 78), 256, 0, stream>>>(xdpre, xd, (long long)Bg * 78);

  // ---- final MLP ----------------------------------------------------------
  assemble_xc_k<<<gsz((long long)Bg * 1792), 256, 0, stream>>>(pg, xd, TGT, xcbf, (long long)Bg * 1792);
  gemm(xcbf, wC1, FC1B, 1, mlpA, Bg, 1024, 1792);
  packA(mlpA, xcbf, Bg, 1024, 1024, 1024);
  gemm(xcbf, wC2, FC2B, 1, mlpB, Bg, 1024, 1024);
  packA(mlpB, xcbf, Bg, 1024, 1024, 1024);
  gemm(xcbf, wC3, FC3B, 1, mlpA, Bg, 512, 1024);
  packA(mlpA, xcbf, Bg, 512, 512, 512);
  gemm(xcbf, wC4, FC4B, 1, mlpB, Bg, 256, 512);
  packA(mlpB, xcbf, Bg, 256, 256, 256);
  gemm(xcbf, wO, OB, 0, out16, Bg, 1, 256);
  copy_out_k<<<gsz(Bg), 256, 0, stream>>>(out16, OUT, Bg);
}
